// ConstructBaselineSuperEdge_68143951118803
// MI455X (gfx1250) — matrix operand order repaired and checked by reading, not compiled, not run on hardware
//
#include <hip/hip_runtime.h>
#include <hip/hip_bf16.h>

typedef __attribute__((ext_vector_type(16))) __bf16 v16bf;
typedef __attribute__((ext_vector_type(8)))  float  v8f;
typedef __attribute__((ext_vector_type(4)))  float  f32x4;

namespace {
constexpr int BATCH  = 8192;
constexpr int NNEI   = 64;     // NM + ND
constexpr int KDIM   = 512;    // EDGE_F + NODE_F
constexpr int A_STR  = 520;    // bf16 elements per A row in LDS (padded: banks 4L+w, conflict-free)
constexpr int KV_STR = 514;    // f32 per KV row in LDS (banks 2n+h, conflict-free)

constexpr size_t OFF_A   = 0;
constexpr size_t OFF_KV  = OFF_A  + (size_t)NNEI * A_STR * 2;   // 66,560
constexpr size_t OFF_PRE = OFF_KV + (size_t)NNEI * KV_STR * 4;  // +131,584
constexpr size_t OFF_EMB = OFF_PRE + 512 * 4;
constexpr size_t OFF_Q   = OFF_EMB + 256 * 4;
constexpr size_t OFF_S   = OFF_Q   + 256 * 4;
constexpr size_t LDS_BYTES = OFF_S + 64 * 4;                    // ~198 KB < 320 KB/WGP
}

__device__ __forceinline__ unsigned short f2bf(float f) {
  unsigned int u = __float_as_uint(f);
  u += 0x7FFFu + ((u >> 16) & 1u);            // round-to-nearest-even
  return (unsigned short)(u >> 16);
}

// Build bf16 column-major fused weight: WcT[j][i] = (j<256 ? Wk[i][j] : Wv[i][j-256])
__global__ void prep_weights_kernel(const float* __restrict__ Wk,
                                    const float* __restrict__ Wv,
                                    unsigned short* __restrict__ WcT) {
  int idx = blockIdx.x * 256 + threadIdx.x;   // 0 .. 512*512-1
  int j = idx >> 9;
  int i = idx & 511;
  float v = (j < 256) ? Wk[i * 256 + j] : Wv[i * 256 + (j - 256)];
  WcT[(size_t)j * 512 + i] = f2bf(v);
}

__global__ void __launch_bounds__(256)
super_edge_kernel(const float* __restrict__ mnode0, const float* __restrict__ mnode1,
                  const float* __restrict__ dnode0, const float* __restrict__ dnode1,
                  const float* __restrict__ mrel0,  const float* __restrict__ drel0,
                  const float* __restrict__ W_edge, const float* __restrict__ b_edge,
                  const float* __restrict__ Wq,     const float* __restrict__ bq,
                  const float* __restrict__ bk,     const float* __restrict__ bv,
                  const unsigned short* __restrict__ WcT,
                  float* __restrict__ out) {
  extern __shared__ char smem[];
  unsigned short* ldsA  = (unsigned short*)(smem + OFF_A);
  float* ldsKV  = (float*)(smem + OFF_KV);
  float* ldsPre = (float*)(smem + OFF_PRE);
  float* ldsEmb = (float*)(smem + OFF_EMB);
  float* ldsQ   = (float*)(smem + OFF_Q);
  float* ldsS   = (float*)(smem + OFF_S);

  const int b = blockIdx.x;
  const int t = threadIdx.x;

  // ---- Phase 0: stage pre_md and edge_nei (A, bf16) into LDS ----
  for (int idx = t; idx < 512; idx += 256) {
    ldsPre[idx] = (idx < 256) ? mnode0[b * 256 + idx] : dnode0[b * 256 + (idx - 256)];
  }
  for (int e = t; e < NNEI * KDIM; e += 256) {
    int n = e >> 9;
    int i = e & 511;
    float v;
    if (n < 32) {
      v = (i < 256) ? mrel0[((size_t)b * 32 + n) * 256 + i]
                    : mnode1[((size_t)b * 32 + n) * 256 + (i - 256)];
    } else {
      int m = n - 32;
      v = (i < 256) ? drel0[((size_t)b * 32 + m) * 256 + i]
                    : dnode1[((size_t)b * 32 + m) * 256 + (i - 256)];
    }
    ldsA[n * A_STR + i] = f2bf(v);
  }
  __syncthreads();

  // ---- Phase 1: edge_emb = relu(pre_md @ W_edge + b_edge), 1 output/thread ----
  {
    const int j = t;
    float acc = b_edge[j];
    #pragma unroll 4
    for (int i = 0; i < 512; ++i) acc = fmaf(ldsPre[i], W_edge[i * 256 + j], acc);
    acc = fmaxf(acc, 0.0f);
    ldsEmb[j] = acc;
    out[(size_t)b * 512 + j] = acc;
  }
  __syncthreads();

  // ---- Phase 2: q = edge_emb @ Wq + bq ----
  {
    const int h = t;
    float acc = bq[h];
    #pragma unroll 4
    for (int j = 0; j < 256; ++j) acc = fmaf(ldsEmb[j], Wq[j * 256 + h], acc);
    ldsQ[h] = acc;
  }

  // ---- Phase 3: KV[64x512] = A[64x512] @ [Wk|Wv] + [bk|bv], bf16 WMMA ----
  const int lane = t & 31;
  const int wave = t >> 5;
  const int lcol = lane & 15;   // N within tile / row within A tile
  const int sel  = lane >> 4;   // K-half selector per WMMA layouts

  for (int cc = 0; cc < 4; ++cc) {
    const int ct = wave * 4 + cc;              // 32 column tiles across 8 waves
    const int C0 = ct * 16;
    const unsigned short* wt = WcT + (size_t)(C0 + lcol) * 512 + sel * 16;
    if (cc < 3)
      __builtin_prefetch(WcT + (size_t)(C0 + 16 + lcol) * 512, 0, 1);

    v8f acc[4] = {v8f{}, v8f{}, v8f{}, v8f{}};
    #pragma unroll
    for (int k = 0; k < 16; ++k) {
      // B fragment: lane L -> column C0+L, 16 contiguous bf16 K-values (pre-transposed)
      v16bf bfrag = *(const v16bf*)(wt + 32 * k);
      const int ka = 32 * k + sel * 16;
      #pragma unroll
      for (int rt = 0; rt < 4; ++rt) {
        const int row = rt * 16 + lcol;
        union { v16bf bf; f32x4 q[2]; } a;
        a.q[0] = *(const f32x4*)(ldsA + row * A_STR + ka);        // K = 32k+16*sel .. +7
        a.q[1] = *(const f32x4*)(ldsA + row * A_STR + ka + 8);    // K = +8 .. +15
        acc[rt] = __builtin_amdgcn_wmma_f32_16x16x32_bf16(
            false, a.bf, false, bfrag, (short)0, acc[rt], false, false);
      }
    }
    const int col = C0 + lcol;
    const float bias = (col < 256) ? bk[col] : bv[col - 256];
    #pragma unroll
    for (int rt = 0; rt < 4; ++rt) {
      #pragma unroll
      for (int r = 0; r < 8; ++r) {
        const int row = rt * 16 + r + sel * 8;   // C/D layout: M = r (+8 for upper lanes)
        ldsKV[row * KV_STR + col] = acc[rt][r] + bias;
      }
    }
  }
  __syncthreads();

  // ---- Phase 4: scores = q . K^T / 16 ----
  if (t < 64) {
    float acc = 0.0f;
    #pragma unroll 4
    for (int h = 0; h < 256; ++h) acc = fmaf(ldsQ[h], ldsKV[t * KV_STR + h], acc);
    ldsS[t] = acc * 0.0625f;   // 1/sqrt(256)
  }
  __syncthreads();

  // ---- Phase 5: softmax (redundant per-thread stats) + attn @ V ----
  float mx = -3.4e38f;
  #pragma unroll 8
  for (int n = 0; n < 64; ++n) mx = fmaxf(mx, ldsS[n]);
  float se = 0.0f;
  #pragma unroll 8
  for (int n = 0; n < 64; ++n) se += __expf(ldsS[n] - mx);
  const float inv = 1.0f / se;

  {
    const int h = t;
    float o = 0.0f;
    #pragma unroll 4
    for (int n = 0; n < 64; ++n)
      o = fmaf(__expf(ldsS[n] - mx), ldsKV[n * KV_STR + 256 + h], o);
    out[(size_t)b * 512 + 256 + h] = o * inv;
  }
}

extern "C" void kernel_launch(void* const* d_in, const int* in_sizes, int n_in,
                              void* d_out, int out_size, void* d_ws, size_t ws_size,
                              hipStream_t stream) {
  const float* mnode0 = (const float*)d_in[0];
  const float* mnode1 = (const float*)d_in[1];
  const float* dnode0 = (const float*)d_in[2];
  const float* dnode1 = (const float*)d_in[3];
  const float* mrel0  = (const float*)d_in[4];
  const float* drel0  = (const float*)d_in[5];
  const float* W_edge = (const float*)d_in[6];
  const float* b_edge = (const float*)d_in[7];
  const float* Wq     = (const float*)d_in[8];
  const float* bq     = (const float*)d_in[9];
  const float* Wk     = (const float*)d_in[10];
  const float* bk     = (const float*)d_in[11];
  const float* Wv     = (const float*)d_in[12];
  const float* bv     = (const float*)d_in[13];
  float* out = (float*)d_out;

  unsigned short* WcT = (unsigned short*)d_ws;   // 512*512 bf16 = 512 KB scratch

  prep_weights_kernel<<<dim3(1024), dim3(256), 0, stream>>>(Wk, Wv, WcT);
  super_edge_kernel<<<dim3(BATCH), dim3(256), LDS_BYTES, stream>>>(
      mnode0, mnode1, dnode0, dnode1, mrel0, drel0,
      W_edge, b_edge, Wq, bq, bk, bv, WcT, out);
}
